// MultichannelMultiheadAttention_23596550324556
// MI455X (gfx1250) — matrix operand order repaired and checked by reading, not compiled, not run on hardware
//
#include <hip/hip_runtime.h>

typedef __attribute__((ext_vector_type(16))) __bf16 v16bf;
typedef __attribute__((ext_vector_type(8)))  __bf16 v8bf;
typedef __attribute__((ext_vector_type(4)))  __bf16 v4bf;
typedef __attribute__((ext_vector_type(8)))  float  v8f;
typedef __attribute__((ext_vector_type(4)))  float  v4f;
typedef __attribute__((ext_vector_type(4)))  unsigned int v4u;
typedef __attribute__((ext_vector_type(8)))  int    v8i;
typedef __attribute__((ext_vector_type(4)))  int    v4i;

#ifndef __has_builtin
#define __has_builtin(x) 0
#endif
#if __has_builtin(__builtin_amdgcn_tensor_load_to_lds) && \
    __has_builtin(__builtin_amdgcn_s_wait_tensorcnt)
#define USE_TDM 1
#else
#define USE_TDM 0
#endif

__device__ __forceinline__ __bf16 bf(float f) { return (__bf16)f; }  // native v_cvt

// ---------------------------------------------------------------------------
// Kernel 0: f32 -> bf16 prepack (bandwidth-bound, one shot)
// ---------------------------------------------------------------------------
__global__ void __launch_bounds__(256)
cvt_bf16(const float* __restrict__ src, __bf16* __restrict__ dst, int n)
{
    int i = (blockIdx.x * 256 + threadIdx.x) * 4;
    if (i < n) {
        v4f f = *(const v4f*)(src + i);
        v4bf o;
        o[0] = bf(f[0]); o[1] = bf(f[1]); o[2] = bf(f[2]); o[3] = bf(f[3]);
        *(v4bf*)(dst + i) = o;
    }
}

// ---------------------------------------------------------------------------
// Kernel 1: per-channel projection GEMM  Y[f,w] = W[f,:] @ X[:,w] + b[f]
//   X: bf16 [bc][256][1024], W: bf16 [c][256][256], b: f32 [c][256]
//   Y: bf16 [bc][1024][256]  (TRANSPOSED: w-major rows, f contiguous)
// Block = one 16-wide w-tile, 4 waves split f into 4x64. The shared X chunk
// (32K x 16N) is staged transposed in LDS once per K-step, so every fragment
// load (A from global, B from LDS) is contiguous.
// ---------------------------------------------------------------------------
__global__ void __launch_bounds__(128)
proj_gemm(const __bf16* __restrict__ X, const __bf16* __restrict__ W,
          const float* __restrict__ B, __bf16* __restrict__ Y)
{
    __shared__ __align__(16) __bf16 Xt[16][32];   // [w][h] transposed chunk

    const int tid  = threadIdx.x;
    const int lane = tid & 31;
    const int wid  = tid >> 5;
    const int ln   = lane & 15;
    const int lh   = lane >> 4;
    const int bc   = blockIdx.z;
    const int c    = bc & 7;
    const int wb   = blockIdx.x * 16;
    const int fb   = wid * 64;
    const int trow = tid >> 2;          // staging: h row 0..31
    const int tseg = (tid & 3) * 4;     // staging: w segment 0/4/8/12

    const __bf16* Wp = W + (size_t)c  * 256 * 256;
    const __bf16* Xp = X + (size_t)bc * 256 * 1024;

    v8f acc[4] = {};
    for (int kk = 0; kk < 256; kk += 32) {
        // stage 32h x 16w chunk transposed into LDS (coalesced b64 reads)
        {
            v4bf xv = *(const v4bf*)(Xp + (size_t)(kk + trow) * 1024 + wb + tseg);
            #pragma unroll
            for (int e = 0; e < 4; ++e)
                Xt[tseg + e][trow] = xv[e];
        }
        __syncthreads();
        // B fragment (shared by all 4 waves): contiguous LDS reads
        union { v16bf v; v8bf h[2]; } b;
        {
            const __bf16* p = &Xt[ln][lh * 8];
            b.h[0] = *(const v8bf*)p;  b.h[1] = *(const v8bf*)(p + 16);
        }
        // 4 A fragments (contiguous b128 global loads), 4 WMMAs
        #pragma unroll
        for (int t = 0; t < 4; ++t) {
            union { v16bf v; v8bf h[2]; } a;
            const __bf16* wr = Wp + (size_t)(fb + t * 16 + ln) * 256 + kk;
            a.h[0] = *(const v8bf*)(wr + lh * 8);
            a.h[1] = *(const v8bf*)(wr + 16 + lh * 8);
            acc[t] = __builtin_amdgcn_wmma_f32_16x16x32_bf16(
                         false, a.v, false, b.v, (short)0, acc[t], false, false);
        }
        __syncthreads();
    }
    // C layout: row M = r + 8*lh (f dim), col N = ln (w dim). Store transposed.
    __bf16* yp = Y + (size_t)bc * 1024 * 256 + (size_t)(wb + ln) * 256;
    #pragma unroll
    for (int t = 0; t < 4; ++t) {
        const float* bp = B + (size_t)c * 256 + fb + t * 16 + lh * 8;
        #pragma unroll
        for (int r = 0; r < 8; ++r)
            yp[fb + t * 16 + lh * 8 + r] = bf(acc[t][r] + bp[r]);
    }
}

// ---------------------------------------------------------------------------
// Kernel 2: flash attention per (bc, head), one wave per 16 query rows.
// 4 waves share (bc, head): the 32-key K chunk is DMA'd into LDS by the
// Tensor Data Mover (TENSOR_LOAD_TO_LDS, TENSORcnt-tracked) when available;
// V is staged transposed so its B-fragments are contiguous LDS reads.
//   Q/K/V: bf16 [bc][1024][256];  QK: f32 [bc][8][1024][1024] (raw scores)
//   A: bf16 [bc][256][1024] (GEMM-ready for Wo)
// ---------------------------------------------------------------------------
__global__ void __launch_bounds__(128)
attn_flash(const __bf16* __restrict__ Q, const __bf16* __restrict__ K,
           const __bf16* __restrict__ V, float* __restrict__ QK,
           __bf16* __restrict__ A)
{
    __shared__ __align__(16) __bf16 Kl[32][32];      // [key][hd]
    __shared__ __align__(16) __bf16 Vt[32][32];      // [hd][key] (transposed)
    __shared__ __align__(16) __bf16 Pl[4][16][32];   // per-wave P re-layout

    const int tid  = threadIdx.x;
    const int lane = tid & 31;
    const int wid  = tid >> 5;
    const int ln   = lane & 15;
    const int lh   = lane >> 4;
    const int bc   = blockIdx.z;
    const int head = blockIdx.y;
    const int qb   = (blockIdx.x * 4 + wid) * 16;
    const int srow = tid >> 2;          // staging: row 0..31
    const int sseg = (tid & 3) * 8;     // staging: 8-elem segment

    const __bf16* Qp = Q + (size_t)bc * 1024 * 256 + (size_t)head * 32;
    const __bf16* Kp = K + (size_t)bc * 1024 * 256 + (size_t)head * 32;
    const __bf16* Vp = V + (size_t)bc * 1024 * 256 + (size_t)head * 32;
    float* qkp = QK + ((size_t)(bc * 8 + head) * 1024 + qb) * 1024;

    // Q A-fragment (16 q-rows x K=32 = head_dim), contiguous per lane half
    union { v16bf v; v8bf h[2]; } qa;
    {
        const __bf16* p = Qp + (size_t)(qb + ln) * 256;
        qa.h[0] = *(const v8bf*)(p + lh * 8);
        qa.h[1] = *(const v8bf*)(p + 16 + lh * 8);
    }

    float m[8], l[8];
    #pragma unroll
    for (int r = 0; r < 8; ++r) { m[r] = -3.0e38f; l[r] = 0.0f; }
    v8f acc0 = {}, acc1 = {};

    for (int kt = 0; kt < 1024; kt += 32) {
#if USE_TDM
        // ---- K chunk: Tensor Data Mover DMA into LDS (wave 0 issues) ----
        if (wid == 0) {
            unsigned kl_lds = (unsigned)(unsigned long long)(size_t)&Kl[0][0];
            unsigned long long ga =
                (unsigned long long)(size_t)(Kp + (size_t)kt * 256);
            v4u g0 = { 1u,                                    // count=1 (user D#)
                       kl_lds,                                // lds_addr
                       (unsigned)(ga & 0xFFFFFFFFu),          // global_addr lo
                       (unsigned)((ga >> 32) & 0x1FFFFFFu) | (2u << 30) }; // type=2
            v8i g1 = { (int)(1u << 16),                       // data_size = 2B
                       (int)((256u & 0xFFFFu) << 16),         // tensor_dim0 lo16
                       (int)((1024u & 0xFFFFu) << 16),        // dim0 hi | dim1 lo
                       (int)(32u << 16),                      // dim1 hi | tile_dim0=32
                       (int)32,                               // tile_dim1=32
                       (int)256,                              // dim0_stride lo
                       0, 0 };
            v4i gz = { 0, 0, 0, 0 };
#if __clang_major__ >= 23
            v8i gz8 = { 0, 0, 0, 0, 0, 0, 0, 0 };
            __builtin_amdgcn_tensor_load_to_lds(g0, g1, gz, gz, gz8, 0);
#else
            __builtin_amdgcn_tensor_load_to_lds(g0, g1, gz, gz, 0);
#endif
            __builtin_amdgcn_s_wait_tensorcnt(0);
        }
#else
        // ---- K chunk: cooperative staged copy ----
        {
            v8bf kv = *(const v8bf*)(Kp + (size_t)(kt + srow) * 256 + sseg);
            *(v8bf*)&Kl[srow][sseg] = kv;
        }
#endif
        // ---- V chunk: cooperative transposed staging ----
        {
            v8bf vv = *(const v8bf*)(Vp + (size_t)(kt + srow) * 256 + sseg);
            #pragma unroll
            for (int e = 0; e < 8; ++e)
                Vt[sseg + e][srow] = vv[e];
        }
        if (kt + 32 < 1024) {   // prefetch next chunk (global_prefetch_b8)
            __builtin_prefetch(Kp + (size_t)(kt + 32 + srow) * 256 + sseg, 0, 1);
            __builtin_prefetch(Vp + (size_t)(kt + 32 + srow) * 256 + sseg, 0, 1);
        }
        __syncthreads();

        // K^T B-fragments from LDS (contiguous: Kl is [key][hd])
        union { v16bf v; v8bf h[2]; } kb0, kb1;
        {
            const __bf16* p0 = &Kl[ln][lh * 8];
            kb0.h[0] = *(const v8bf*)p0;  kb0.h[1] = *(const v8bf*)(p0 + 16);
            const __bf16* p1 = &Kl[16 + ln][lh * 8];
            kb1.h[0] = *(const v8bf*)p1;  kb1.h[1] = *(const v8bf*)(p1 + 16);
        }
        v8f z = {};
        v8f s0 = __builtin_amdgcn_wmma_f32_16x16x32_bf16(
                     false, qa.v, false, kb0.v, (short)0, z, false, false);
        v8f s1 = __builtin_amdgcn_wmma_f32_16x16x32_bf16(
                     false, qa.v, false, kb1.v, (short)0, z, false, false);
        #pragma unroll
        for (int r = 0; r < 8; ++r) { s0[r] *= 0.0625f; s1[r] *= 0.0625f; }

        // write raw (pre-softmax) scaled scores — second reference output
        #pragma unroll
        for (int r = 0; r < 8; ++r) {
            size_t row = (size_t)(r + 8 * lh) * 1024;
            qkp[row + kt + ln]      = s0[r];
            qkp[row + kt + 16 + ln] = s1[r];
        }

        // online softmax update (row = M = r + 8*lh; cols across 16-lane half)
        #pragma unroll
        for (int r = 0; r < 8; ++r) {
            float t = fmaxf(s0[r], s1[r]);
            t = fmaxf(t, __shfl_xor(t, 1, 32));
            t = fmaxf(t, __shfl_xor(t, 2, 32));
            t = fmaxf(t, __shfl_xor(t, 4, 32));
            t = fmaxf(t, __shfl_xor(t, 8, 32));
            float mnew = fmaxf(m[r], t);
            float corr = __expf(m[r] - mnew);
            float p0 = __expf(s0[r] - mnew);
            float p1 = __expf(s1[r] - mnew);
            float rs = p0 + p1;
            rs += __shfl_xor(rs, 1, 32);
            rs += __shfl_xor(rs, 2, 32);
            rs += __shfl_xor(rs, 4, 32);
            rs += __shfl_xor(rs, 8, 32);
            l[r] = l[r] * corr + rs;
            m[r] = mnew;
            acc0[r] *= corr;
            acc1[r] *= corr;
            Pl[wid][r + 8 * lh][ln]      = bf(p0);
            Pl[wid][r + 8 * lh][16 + ln] = bf(p1);
        }

        // reload P as an A fragment (wave-local buffer; LDS ops are in-order)
        union { v16bf v; v8bf h[2]; } pa;
        {
            const __bf16* pp = &Pl[wid][ln][0];
            pa.h[0] = *(const v8bf*)(pp + lh * 8);
            pa.h[1] = *(const v8bf*)(pp + 16 + lh * 8);
        }
        // V B-fragments from transposed LDS tile (contiguous)
        union { v16bf v; v8bf h[2]; } vb0, vb1;
        {
            const __bf16* q0 = &Vt[ln][lh * 8];
            vb0.h[0] = *(const v8bf*)q0;  vb0.h[1] = *(const v8bf*)(q0 + 16);
            const __bf16* q1 = &Vt[16 + ln][lh * 8];
            vb1.h[0] = *(const v8bf*)q1;  vb1.h[1] = *(const v8bf*)(q1 + 16);
        }
        acc0 = __builtin_amdgcn_wmma_f32_16x16x32_bf16(
                   false, pa.v, false, vb0.v, (short)0, acc0, false, false);
        acc1 = __builtin_amdgcn_wmma_f32_16x16x32_bf16(
                   false, pa.v, false, vb1.v, (short)0, acc1, false, false);
        __syncthreads();   // guard Kl/Vt against next iteration's staging
    }

    // normalize, store A as [f = head*32+hd][w = qpos] for the output GEMM
    __bf16* ap = A + (size_t)bc * 256 * 1024;
    #pragma unroll
    for (int r = 0; r < 8; ++r) {
        int qrow = qb + r + 8 * lh;
        float inv = 1.0f / l[r];
        ap[(size_t)(head * 32 + ln) * 1024 + qrow]      = bf(acc0[r] * inv);
        ap[(size_t)(head * 32 + 16 + ln) * 1024 + qrow] = bf(acc1[r] * inv);
    }
}

// ---------------------------------------------------------------------------
// Kernel 3: output projection  Out[f,w] = Wo[f,:] @ A[:,w] + bo[f] (f32 out)
// Same LDS-staged structure as proj_gemm.
// ---------------------------------------------------------------------------
__global__ void __launch_bounds__(128)
out_proj(const __bf16* __restrict__ Aws, const __bf16* __restrict__ W,
         const float* __restrict__ B, float* __restrict__ Out)
{
    __shared__ __align__(16) __bf16 At[16][32];   // [w][h] transposed chunk

    const int tid  = threadIdx.x;
    const int lane = tid & 31;
    const int wid  = tid >> 5;
    const int ln   = lane & 15;
    const int lh   = lane >> 4;
    const int bc   = blockIdx.z;
    const int c    = bc & 7;
    const int wb   = blockIdx.x * 16;
    const int fb   = wid * 64;
    const int trow = tid >> 2;
    const int tseg = (tid & 3) * 4;

    const __bf16* Wp = W   + (size_t)c  * 256 * 256;
    const __bf16* Ap = Aws + (size_t)bc * 256 * 1024;

    v8f acc[4] = {};
    for (int kk = 0; kk < 256; kk += 32) {
        {
            v4bf xv = *(const v4bf*)(Ap + (size_t)(kk + trow) * 1024 + wb + tseg);
            #pragma unroll
            for (int e = 0; e < 4; ++e)
                At[tseg + e][trow] = xv[e];
        }
        __syncthreads();
        union { v16bf v; v8bf h[2]; } b;
        {
            const __bf16* p = &At[ln][lh * 8];
            b.h[0] = *(const v8bf*)p;  b.h[1] = *(const v8bf*)(p + 16);
        }
        #pragma unroll
        for (int t = 0; t < 4; ++t) {
            union { v16bf v; v8bf h[2]; } a;
            const __bf16* wr = Wp + (size_t)(fb + t * 16 + ln) * 256 + kk;
            a.h[0] = *(const v8bf*)(wr + lh * 8);
            a.h[1] = *(const v8bf*)(wr + 16 + lh * 8);
            acc[t] = __builtin_amdgcn_wmma_f32_16x16x32_bf16(
                         false, a.v, false, b.v, (short)0, acc[t], false, false);
        }
        __syncthreads();
    }
    #pragma unroll
    for (int t = 0; t < 4; ++t) {
        const float* bp = B + (size_t)c * 256 + fb + t * 16 + lh * 8;
        float* op = Out + (size_t)bc * 256 * 1024
                        + (size_t)(fb + t * 16 + lh * 8) * 1024 + wb + ln;
        #pragma unroll
        for (int r = 0; r < 8; ++r)
            op[(size_t)r * 1024] = acc[t][r] + bp[r];
    }
}

// ---------------------------------------------------------------------------
extern "C" void kernel_launch(void* const* d_in, const int* in_sizes, int n_in,
                              void* d_out, int out_size, void* d_ws, size_t ws_size,
                              hipStream_t stream)
{
    const float* x  = (const float*)d_in[0];
    const float* Wq = (const float*)d_in[1];
    const float* bq = (const float*)d_in[2];
    const float* Wk = (const float*)d_in[3];
    const float* bk = (const float*)d_in[4];
    const float* Wv = (const float*)d_in[5];
    const float* bv = (const float*)d_in[6];
    const float* Wo = (const float*)d_in[7];
    const float* bo = (const float*)d_in[8];

    float* out = (float*)d_out;                       // (2,8,256,1024)
    float* qk  = out + (size_t)2 * 8 * 256 * 1024;    // (2,8,8,1024,1024)

    const size_t NX = (size_t)16 * 256 * 1024;        // x / qkva plane elems
    const size_t NW = (size_t)8 * 256 * 256;          // one weight tensor elems

    __bf16* xbf = (__bf16*)d_ws;
    __bf16* wqb = xbf + NX;
    __bf16* wkb = wqb + NW;
    __bf16* wvb = wkb + NW;
    __bf16* wob = wvb + NW;
    __bf16* qws = wob + NW;
    __bf16* kws = qws + NX;
    __bf16* vws = kws + NX;
    __bf16* aws = vws + NX;

    // prepack everything to bf16 once (bandwidth-trivial)
    cvt_bf16<<<dim3((unsigned)(NX / 4 / 256)), dim3(256), 0, stream>>>(x,  xbf, (int)NX);
    cvt_bf16<<<dim3((unsigned)(NW / 4 / 256)), dim3(256), 0, stream>>>(Wq, wqb, (int)NW);
    cvt_bf16<<<dim3((unsigned)(NW / 4 / 256)), dim3(256), 0, stream>>>(Wk, wkb, (int)NW);
    cvt_bf16<<<dim3((unsigned)(NW / 4 / 256)), dim3(256), 0, stream>>>(Wv, wvb, (int)NW);
    cvt_bf16<<<dim3((unsigned)(NW / 4 / 256)), dim3(256), 0, stream>>>(Wo, wob, (int)NW);

    dim3 blk(128, 1, 1);
    dim3 gproj(64, 1, 16);    // w-tiles, -, bc  (4 waves split f into 4x64)
    proj_gemm<<<gproj, blk, 0, stream>>>(xbf, wqb, bq, qws);
    proj_gemm<<<gproj, blk, 0, stream>>>(xbf, wkb, bk, kws);
    proj_gemm<<<gproj, blk, 0, stream>>>(xbf, wvb, bv, vws);

    dim3 gattn(16, 8, 16);    // qtile-groups(x4 waves), heads, bc
    attn_flash<<<gattn, blk, 0, stream>>>(qws, kws, vws, qk, aws);

    out_proj<<<gproj, blk, 0, stream>>>(aws, wob, bo, out);
}
